// MaskedSelfRelation_44375602102656
// MI455X (gfx1250) — matrix operand order repaired and checked by reading, not compiled, not run on hardware
//
#include <hip/hip_runtime.h>
#include <hip/hip_bf16.h>

// ---------------------------------------------------------------------------
// MaskedSelfRelation for MI455X (gfx1250, wave32, WMMA + async-to-LDS)
//
//   H=768, R=64 heads, DH=12, B=4, S=512.
//   rel[b,i,j,t] = sum_r Wt[t,r]*q[i,r]*k[j,r]  ==> head t uses Q' = q .* Wt[t,:]
//   => 64-head attention, shared K (dim 64), per-head V (dim 12).
// ---------------------------------------------------------------------------

typedef __attribute__((ext_vector_type(16))) _Float16 v16h;
typedef __attribute__((ext_vector_type(8)))  _Float16 v8h;
typedef __attribute__((ext_vector_type(8)))  float    v8f;

#define H_DIM 768
#define R_DIM 64
#define DH_DIM 12
#define B_DIM 4
#define S_DIM 512
#define ROWS (B_DIM * S_DIM)            // 2048
#define NPROJ (R_DIM + R_DIM + H_DIM)   // 896

static __device__ __forceinline__ v8f wmma_f16(v16h a, v16h b, v8f c) {
    return __builtin_amdgcn_wmma_f32_16x16x32_f16(false, a, false, b, (short)0, c,
                                                  false, false);
}

// A-fragment K index map for 16-bit 16x32 A operand (contiguous, matches B):
// lanes 0-15 : halves 0..15 -> K 0..15
// lanes16-31 : halves 0..15 -> K 16..31
static __device__ __forceinline__ int a_kmap(int h, int laneHi) {
    return h + (laneHi ? 16 : 0);
}

// cross-lane row reductions within 16-lane halves (D-layout: N = lane%16)
static __device__ __forceinline__ float rmax16(float v) {
#pragma unroll
    for (int m = 1; m < 16; m <<= 1) v = fmaxf(v, __shfl_xor(v, m, 16));
    return v;
}
static __device__ __forceinline__ float rsum16(float v) {
#pragma unroll
    for (int m = 1; m < 16; m <<= 1) v += __shfl_xor(v, m, 16);
    return v;
}

// ---------------------------------------------------------------------------
// Kernel 0: f32 -> f16 conversion (weights)
// ---------------------------------------------------------------------------
__global__ __launch_bounds__(256) void cvt_f32_f16(const float* __restrict__ src,
                                                   _Float16* __restrict__ dst, int n) {
    int i = blockIdx.x * 256 + threadIdx.x;
    if (i < n) dst[i] = (_Float16)src[i];
}

// ---------------------------------------------------------------------------
// Kernel 1: LayerNorm -> f16 activations. One block per row.
// ---------------------------------------------------------------------------
__global__ __launch_bounds__(256) void ln_kernel(const float* __restrict__ x,
                                                 const float* __restrict__ g,
                                                 const float* __restrict__ bb,
                                                 _Float16* __restrict__ xn) {
    __shared__ float red[256];
    const int row = blockIdx.x, tid = threadIdx.x;
    const float* xr = x + (size_t)row * H_DIM;

    float s = 0.f;
    for (int c = tid; c < H_DIM; c += 256) s += xr[c];
    red[tid] = s; __syncthreads();
#pragma unroll
    for (int off = 128; off > 0; off >>= 1) {
        if (tid < off) red[tid] += red[tid + off];
        __syncthreads();
    }
    const float mu = red[0] * (1.f / H_DIM);
    __syncthreads();

    float v = 0.f;
    for (int c = tid; c < H_DIM; c += 256) { float d = xr[c] - mu; v += d * d; }
    red[tid] = v; __syncthreads();
#pragma unroll
    for (int off = 128; off > 0; off >>= 1) {
        if (tid < off) red[tid] += red[tid + off];
        __syncthreads();
    }
    const float rstd = rsqrtf(red[0] * (1.f / H_DIM) + 1e-5f);

    for (int c = tid; c < H_DIM; c += 256)
        xn[(size_t)row * H_DIM + c] = (_Float16)((xr[c] - mu) * rstd * g[c] + bb[c]);
}

// ---------------------------------------------------------------------------
// Kernel 2: projection GEMM  C[2048,896] = xn_h[2048,768] @ Wcat_cm.
// One wave computes a 16x64 strip (4 N-subtiles share one A fragment).
// Outputs: q f32 [2048,64], k f16 [2048,64], vT f16 [B,64,12,S] (bias folded).
// ---------------------------------------------------------------------------
__global__ __launch_bounds__(32) void proj_gemm(const _Float16* __restrict__ A,
                                                const _Float16* __restrict__ Bw,
                                                const float* __restrict__ bq,
                                                const float* __restrict__ bk,
                                                const float* __restrict__ bv,
                                                float* __restrict__ qf,
                                                _Float16* __restrict__ kh,
                                                _Float16* __restrict__ vT) {
    const int lane = threadIdx.x, n16 = lane & 15, laneHi = lane >> 4;
    const int tm = blockIdx.x, tn4 = blockIdx.y;

    v8f c[4];
    const _Float16* brow[4];
#pragma unroll
    for (int s = 0; s < 4; ++s) {
        const int n = tn4 * 64 + s * 16 + n16;
        const float bias = (n < R_DIM) ? bq[n]
                         : (n < 2 * R_DIM) ? bk[n - R_DIM]
                                           : bv[n - 2 * R_DIM];
#pragma unroll
        for (int g = 0; g < 8; ++g) c[s][g] = bias;
        brow[s] = Bw + (size_t)n * H_DIM;
    }

    const _Float16* arow = A + (size_t)(tm * 16 + n16) * H_DIM;

    for (int kb = 0; kb < H_DIM; kb += 32) {
        if (kb + 128 < H_DIM) {
            __builtin_prefetch(arow + kb + 128, 0, 3);
            __builtin_prefetch(brow[0] + kb + 128, 0, 3);
        }
        v8h a0 = *(const v8h*)(arow + kb + (laneHi ? 16 : 0));
        v8h a1 = *(const v8h*)(arow + kb + 8 + (laneHi ? 16 : 0));
        v16h av = __builtin_shufflevector(a0, a1, 0, 1, 2, 3, 4, 5, 6, 7,
                                          8, 9, 10, 11, 12, 13, 14, 15);
#pragma unroll
        for (int s = 0; s < 4; ++s) {
            v16h bvv = *(const v16h*)(brow[s] + kb + (laneHi ? 16 : 0));
            c[s] = wmma_f16(av, bvv, c[s]);
        }
    }

#pragma unroll
    for (int s = 0; s < 4; ++s) {
        const int n = tn4 * 64 + s * 16 + n16;
#pragma unroll
        for (int g = 0; g < 8; ++g) {
            const int m = tm * 16 + g + 8 * laneHi;
            if (n < R_DIM) {
                qf[(size_t)m * R_DIM + n] = c[s][g];
            } else if (n < 2 * R_DIM) {
                kh[(size_t)m * R_DIM + (n - R_DIM)] = (_Float16)c[s][g];
            } else {
                const int t = (n - 2 * R_DIM) / DH_DIM, d = (n - 2 * R_DIM) % DH_DIM;
                const int bidx = m >> 9, srow = m & (S_DIM - 1);
                vT[(((size_t)bidx * R_DIM + t) * DH_DIM + d) * S_DIM + srow] =
                    (_Float16)c[s][g];
            }
        }
    }
}

// ---------------------------------------------------------------------------
// Kernel 3: flash attention. grid = (S/64, 64 heads, 4 batches), 4 waves/WG.
// Each 32x64 K tile is staged into LDS ONCE per workgroup with CDNA5
// global_load_async_to_lds_b128 (ASYNCcnt), shared by all 4 q-tile waves;
// next tile's async issue overlaps softmax + P@V of the current tile.
// ---------------------------------------------------------------------------
__global__ __launch_bounds__(128) void attn_kernel(const float* __restrict__ qf,
                                                   const _Float16* __restrict__ kh,
                                                   const _Float16* __restrict__ vT,
                                                   const float* __restrict__ Wt,
                                                   const float* __restrict__ mask,
                                                   _Float16* __restrict__ oh) {
    const int tid = threadIdx.x;
    const int lane = tid & 31, n16 = lane & 15, laneHi = (lane >> 4) & 1;
    const int wave = tid >> 5;
    const int t = blockIdx.y, b = blockIdx.z;
    const int qt = blockIdx.x * 4 + wave;

    __shared__ _Float16 ktile[32 * R_DIM];     // 4 KB: K block [32 j][64 r]
    __shared__ _Float16 pbuf[4 * 16 * 32];     // 4 KB: per-wave P transpose tile
    _Float16* pw = pbuf + wave * 16 * 32;

    // ---- Q' A-fragments: Q'[i,r] = q[i,r] * Wt[t,r]
    const int rowm = b * S_DIM + qt * 16 + n16;
    v16h aQ0, aQ1;
#pragma unroll
    for (int h = 0; h < 16; ++h) {
        const int r = a_kmap(h, laneHi);
        aQ0[h] = (_Float16)(qf[(size_t)rowm * R_DIM + r] * Wt[t * R_DIM + r]);
        aQ1[h] = (_Float16)(qf[(size_t)rowm * R_DIM + r + 32] * Wt[t * R_DIM + r + 32]);
    }

    v8f o = {};
    float mrow[8], lrow[8];
#pragma unroll
    for (int g = 0; g < 8; ++g) { mrow[g] = -1e30f; lrow[g] = 0.f; }

    // ---- async K-tile stage: 32 rows x 128 B = 256 x 16 B, 2 per thread
    auto issue_k = [&](int jb) {
#pragma unroll
        for (int e = 0; e < 2; ++e) {
            const int idx = tid + e * 128;        // 0..255
            const int j = idx >> 3, cch = idx & 7;
            const _Float16* src = kh + ((size_t)(b * S_DIM + jb + j) << 6) + cch * 8;
            const unsigned lds = (unsigned)(uintptr_t)(ktile + (j << 6) + cch * 8);
            asm volatile("global_load_async_to_lds_b128 %0, %1, off"
                         :: "v"(lds), "v"(src) : "memory");
        }
    };

    issue_k(0);

    for (int jb = 0; jb < S_DIM; jb += 32) {
        asm volatile("s_wait_asynccnt 0x0" ::: "memory");
        __syncthreads();   // K tile resident in LDS

        // ---- scores: two 16x16 D tiles from LDS K, K(=r) dim 64
        v8f s0 = {}, s1 = {};
        {
            const _Float16* kr = ktile + ((0 * 16 + n16) << 6) + laneHi * 16;
            v16h b0 = *(const v16h*)(kr);
            v16h b1 = *(const v16h*)(kr + 32);
            s0 = wmma_f16(aQ0, b0, s0);
            s0 = wmma_f16(aQ1, b1, s0);
        }
        {
            const _Float16* kr = ktile + ((1 * 16 + n16) << 6) + laneHi * 16;
            v16h b0 = *(const v16h*)(kr);
            v16h b1 = *(const v16h*)(kr + 32);
            s1 = wmma_f16(aQ0, b0, s1);
            s1 = wmma_f16(aQ1, b1, s1);
        }
        __syncthreads();   // everyone done reading ktile
        if (jb + 32 < S_DIM) issue_k(jb + 32);   // overlap next stage

        // ---- additive mask (indexed by (b, j)), online softmax
        const float mk0 = mask[b * S_DIM + jb + n16];
        const float mk1 = mask[b * S_DIM + jb + 16 + n16];
        float mnew[8], alpha[8];
#pragma unroll
        for (int g = 0; g < 8; ++g) {
            s0[g] += mk0; s1[g] += mk1;
            float rm = rmax16(fmaxf(s0[g], s1[g]));
            mnew[g] = fmaxf(mrow[g], rm);
            alpha[g] = __expf(mrow[g] - mnew[g]);
            mrow[g] = mnew[g];
        }
#pragma unroll
        for (int g = 0; g < 8; ++g) {
            float p0 = __expf(s0[g] - mnew[g]);
            float p1 = __expf(s1[g] - mnew[g]);
            lrow[g] = lrow[g] * alpha[g] + rsum16(p0 + p1);
            o[g] *= alpha[g];
            const int prow = g + 8 * laneHi;
            pw[prow * 32 + n16]      = (_Float16)p0;
            pw[prow * 32 + 16 + n16] = (_Float16)p1;
        }

        // ---- P: D-layout -> A-layout via per-wave LDS tile (in-order DS)
        v16h pa;
#pragma unroll
        for (int h = 0; h < 16; ++h) pa[h] = pw[n16 * 32 + a_kmap(h, laneHi)];

        // ---- V B-fragment: vT[b,t,d,j] makes this one contiguous 32B load
        v16h vb = {};
        if (n16 < DH_DIM) {
            const _Float16* vr =
                vT + (((size_t)b * R_DIM + t) * DH_DIM + n16) * S_DIM + jb + laneHi * 16;
            vb = *(const v16h*)vr;
        }
        o = wmma_f16(pa, vb, o);
    }

    // ---- normalize and store: col d = lane%16 (<12), rows g + 8*laneHi
    if (n16 < DH_DIM) {
#pragma unroll
        for (int g = 0; g < 8; ++g) {
            const int i = qt * 16 + g + 8 * laneHi;
            oh[(size_t)(b * S_DIM + i) * H_DIM + t * DH_DIM + n16] =
                (_Float16)(o[g] / lrow[g]);
        }
    }
}

// ---------------------------------------------------------------------------
// Kernel 4: output projection  out[2048,768] = oh @ Wo^T + bo  (f32 out)
// One wave computes a 16x64 strip.
// ---------------------------------------------------------------------------
__global__ __launch_bounds__(32) void out_gemm(const _Float16* __restrict__ A,
                                               const _Float16* __restrict__ Bw,
                                               const float* __restrict__ bo,
                                               float* __restrict__ out) {
    const int lane = threadIdx.x, n16 = lane & 15, laneHi = lane >> 4;
    const int tm = blockIdx.x, tn4 = blockIdx.y;

    v8f c[4];
    const _Float16* brow[4];
#pragma unroll
    for (int s = 0; s < 4; ++s) {
        const int n = tn4 * 64 + s * 16 + n16;
        const float bias = bo[n];
#pragma unroll
        for (int g = 0; g < 8; ++g) c[s][g] = bias;
        brow[s] = Bw + (size_t)n * H_DIM;
    }

    const _Float16* arow = A + (size_t)(tm * 16 + n16) * H_DIM;

    for (int kb = 0; kb < H_DIM; kb += 32) {
        if (kb + 128 < H_DIM) {
            __builtin_prefetch(arow + kb + 128, 0, 3);
            __builtin_prefetch(brow[0] + kb + 128, 0, 3);
        }
        v8h a0 = *(const v8h*)(arow + kb + (laneHi ? 16 : 0));
        v8h a1 = *(const v8h*)(arow + kb + 8 + (laneHi ? 16 : 0));
        v16h av = __builtin_shufflevector(a0, a1, 0, 1, 2, 3, 4, 5, 6, 7,
                                          8, 9, 10, 11, 12, 13, 14, 15);
#pragma unroll
        for (int s = 0; s < 4; ++s) {
            v16h bvv = *(const v16h*)(brow[s] + kb + (laneHi ? 16 : 0));
            c[s] = wmma_f16(av, bvv, c[s]);
        }
    }

#pragma unroll
    for (int s = 0; s < 4; ++s) {
        const int n = tn4 * 64 + s * 16 + n16;
#pragma unroll
        for (int g = 0; g < 8; ++g) {
            const int m = tm * 16 + g + 8 * laneHi;
            out[(size_t)m * H_DIM + n] = c[s][g];
        }
    }
}

// ---------------------------------------------------------------------------
extern "C" void kernel_launch(void* const* d_in, const int* in_sizes, int n_in,
                              void* d_out, int out_size, void* d_ws, size_t ws_size,
                              hipStream_t stream) {
    const float* tgt  = (const float*)d_in[0];
    const float* mask = (const float*)d_in[1];
    const float* ln_g = (const float*)d_in[2];
    const float* ln_b = (const float*)d_in[3];
    const float* Wq   = (const float*)d_in[4];
    const float* bq   = (const float*)d_in[5];
    const float* Wk   = (const float*)d_in[6];
    const float* bk   = (const float*)d_in[7];
    const float* Wv   = (const float*)d_in[8];
    const float* bv   = (const float*)d_in[9];
    const float* Wt   = (const float*)d_in[10];
    const float* Wo   = (const float*)d_in[11];
    const float* bo   = (const float*)d_in[12];
    float* out = (float*)d_out;

    // workspace carve-up (bytes)
    char* ws = (char*)d_ws;
    _Float16* xnh  = (_Float16*)(ws);                                   // 2048*768
    _Float16* wcat = (_Float16*)(ws + 3145728);                         // 896*768
    _Float16* woh  = (_Float16*)(ws + 4521984);                         // 768*768
    float*    qf   = (float*)   (ws + 5701632);                         // 2048*64 f32
    _Float16* kh   = (_Float16*)(ws + 6225920);                         // 2048*64
    _Float16* vT   = (_Float16*)(ws + 6488064);                         // [B,64,12,S]
    _Float16* oh   = (_Float16*)(ws + 9633792);                         // 2048*768

    // 0) weights -> f16 (Wcat rows: [Wq | Wk | Wv] = column-major B operand)
    cvt_f32_f16<<<(R_DIM * H_DIM + 255) / 256, 256, 0, stream>>>(Wq, wcat, R_DIM * H_DIM);
    cvt_f32_f16<<<(R_DIM * H_DIM + 255) / 256, 256, 0, stream>>>(Wk, wcat + R_DIM * H_DIM,
                                                                 R_DIM * H_DIM);
    cvt_f32_f16<<<(H_DIM * H_DIM + 255) / 256, 256, 0, stream>>>(Wv, wcat + 2 * R_DIM * H_DIM,
                                                                 H_DIM * H_DIM);
    cvt_f32_f16<<<(H_DIM * H_DIM + 255) / 256, 256, 0, stream>>>(Wo, woh, H_DIM * H_DIM);

    // 1) LayerNorm
    ln_kernel<<<ROWS, 256, 0, stream>>>(tgt, ln_g, ln_b, xnh);

    // 2) fused q/k/v projection GEMM (WMMA, 16x64 strips)
    proj_gemm<<<dim3(ROWS / 16, NPROJ / 64), 32, 0, stream>>>(xnh, wcat, bq, bk, bv,
                                                              qf, kh, vT);

    // 3) 64-head flash attention (WMMA + async K staging into LDS)
    attn_kernel<<<dim3(S_DIM / 64, R_DIM, B_DIM), 128, 0, stream>>>(qf, kh, vT, Wt,
                                                                    mask, oh);

    // 4) output projection GEMM (WMMA, 16x64 strips) -> f32 out
    out_gemm<<<dim3(ROWS / 16, H_DIM / 64), 32, 0, stream>>>(oh, woh, bo, out);
}
